// MultiHeadAttention_9225589752264
// MI455X (gfx1250) — compile-verified, matrix operand order checked by reading, not run on hardware
//
#include <hip/hip_runtime.h>
#include <hip/hip_bf16.h>

// ---------------------------------------------------------------------------
// MI455X (gfx1250) multi-head attention forward.
//   B=2, S=2048, E=1024, H=16, hd=64
//   d_out = out (2*2048*1024 f32) ++ attn (2*16*2048*2048 f32)
// All matmuls on v_wmma_f32_16x16x32_f16 (f32 accumulate). f32 inputs are
// pre-converted to f16 once; all GEMM fragments are pure b128 loads of packed
// halves. Attention fuses scores+softmax+attn-write+PV with the 16x2048 score
// stripe resident in LDS (CDNA5: 320KB/WGP).
// ---------------------------------------------------------------------------

typedef __attribute__((ext_vector_type(16))) _Float16 v16h;
typedef __attribute__((ext_vector_type(8)))  _Float16 v8h;
typedef __attribute__((ext_vector_type(4)))  _Float16 v4h;
typedef __attribute__((ext_vector_type(8)))  float    v8f;

#define WMMA_F16(A, B, C) \
  __builtin_amdgcn_wmma_f32_16x16x32_f16(false, (A), false, (B), (short)0, (C), false, false)

// 16x32 16-bit A/B fragment per documented CDNA5 layout:
// lane L holds row/col (L&15); K elems {koff..koff+7} U {16+koff..16+koff+7},
// koff = (L>=16)*8.  Two b128 loads of packed halves, no conversion.
__device__ __forceinline__ v16h load_frag(const _Float16* __restrict__ src,
                                          size_t ld, int lane) {
  const int row  = lane & 15;
  const int koff = (lane >> 4) << 3;
  const _Float16* p = src + (size_t)row * ld + koff;
  v8h lo = *(const v8h*)(p);
  v8h hi = *(const v8h*)(p + 16);
  return __builtin_shufflevector(lo, hi, 0, 1, 2, 3, 4, 5, 6, 7,
                                 8, 9, 10, 11, 12, 13, 14, 15);
}

// ---------------------------------------------------------------------------
// Streaming f32 -> f16 convert (vectorized: b128 in, b64 out).
// ---------------------------------------------------------------------------
__global__ __launch_bounds__(256)
void cvt_f32_to_f16(const float* __restrict__ src, _Float16* __restrict__ dst,
                    int n4) {
  const int i = blockIdx.x * 256 + threadIdx.x;
  if (i < n4) {
    const float4 v = ((const float4*)src)[i];
    v4h h;
    h[0] = (_Float16)v.x;
    h[1] = (_Float16)v.y;
    h[2] = (_Float16)v.z;
    h[3] = (_Float16)v.w;
    *(v4h*)(dst + (size_t)i * 4) = h;
  }
}

// ---------------------------------------------------------------------------
// GEMM: C[M,N] = A[M,K] * B[N,K]^T + bias[N]   (einsum 'mk,nk->mn'), f16 in.
// 256 threads = 8 waves; wave tile 64x64 (4x4 WMMA accums), block 256x128.
// One B fragment live at a time caps pressure (~180 VGPRs, no spills).
// ---------------------------------------------------------------------------
template <typename TC>
__global__ __launch_bounds__(256, 1)
void gemm_wmma(const _Float16* __restrict__ A, const _Float16* __restrict__ B,
               const float* __restrict__ bias, TC* __restrict__ C,
               int M, int N, int K) {
  const int tid  = threadIdx.x;
  const int lane = tid & 31;
  const int wave = tid >> 5;
  const int wm   = wave >> 1;          // 0..3
  const int wn   = wave & 1;           // 0..1
  const int row0 = blockIdx.x * 256 + wm * 64;
  const int col0 = blockIdx.y * 128 + wn * 64;

  v8f acc[4][4] = {};

  for (int k0 = 0; k0 < K; k0 += 32) {
    if (k0 + 32 < K)  // gfx1250 global_prefetch path
      __builtin_prefetch(A + (size_t)(row0 + lane) * K + k0 + 32, 0, 3);
    v16h af[4];
#pragma unroll
    for (int i = 0; i < 4; ++i)
      af[i] = load_frag(A + (size_t)(row0 + i * 16) * K + k0, (size_t)K, lane);
#pragma unroll
    for (int j = 0; j < 4; ++j) {
      const v16h bf = load_frag(B + (size_t)(col0 + j * 16) * K + k0, (size_t)K, lane);
#pragma unroll
      for (int i = 0; i < 4; ++i)
        acc[i][j] = WMMA_F16(af[i], bf, acc[i][j]);
    }
  }

  // Epilogue: C/D layout — VGPR r, lane L: row = r + 8*(L>=16), col = L&15.
  const int cn    = lane & 15;
  const int rbase = (lane >> 4) << 3;
#pragma unroll
  for (int i = 0; i < 4; ++i) {
#pragma unroll
    for (int j = 0; j < 4; ++j) {
      const int col = col0 + j * 16 + cn;
      const float bv = bias[col];
#pragma unroll
      for (int r = 0; r < 8; ++r) {
        const int rowm = row0 + i * 16 + rbase + r;
        C[(size_t)rowm * N + col] = (TC)(acc[i][j][r] + bv);
      }
    }
  }
}

// ---------------------------------------------------------------------------
// Fused attention: one block per (b, h, 16-row q tile). 256 threads = 8 waves.
//   qkv (f16) layout: [B*S, 3072]; head h: q at h*192, k at +64, v at +128.
//   LDS: 16x2048 f32 score stripe (128KB) + 8x1024 f32 split-K partials (32KB)
// ---------------------------------------------------------------------------
__global__ __launch_bounds__(256)
void attn_fused(const _Float16* __restrict__ qkv,
                float* __restrict__ attn_out,
                _Float16* __restrict__ vals) {
  extern __shared__ float smem[];
  float* sc   = smem;              // [16][2048] scores / attn probs
  float* part = smem + 16 * 2048;  // [8][16*64] PV split-K partials

  const int tid  = threadIdx.x;
  const int lane = tid & 31;
  const int wave = tid >> 5;

  const int blk = blockIdx.x;      // b*2048 + h*128 + qt
  const int qt  = blk & 127;
  const int h   = (blk >> 7) & 15;
  const int b   = blk >> 11;
  const int q0  = qt * 16;

  const size_t ldq = 3072;
  const _Float16* qbase = qkv + ((size_t)b * 2048 + q0) * ldq + h * 192;
  const _Float16* kbase = qkv + ((size_t)b * 2048) * ldq + h * 192 + 64;
  const _Float16* vbase = qkv + ((size_t)b * 2048) * ldq + h * 192 + 128;

  const int cn    = lane & 15;
  const int rbase = (lane >> 4) << 3;

  // ---- Stage 1: S = (Q K^T)/8; wave w covers k in [w*256, (w+1)*256)
  const v16h aq0 = load_frag(qbase, ldq, lane);        // d = 0..31
  const v16h aq1 = load_frag(qbase + 32, ldq, lane);   // d = 32..63
  for (int t = 0; t < 16; ++t) {
    const int kb = wave * 256 + t * 16;
    const v16h bk0 = load_frag(kbase + (size_t)kb * ldq, ldq, lane);
    const v16h bk1 = load_frag(kbase + (size_t)kb * ldq + 32, ldq, lane);
    v8f c = {};
    c = WMMA_F16(aq0, bk0, c);
    c = WMMA_F16(aq1, bk1, c);
    const int col = kb + cn;
#pragma unroll
    for (int r = 0; r < 8; ++r)
      sc[(rbase + r) * 2048 + col] = c[r] * 0.125f;   // 1/sqrt(64)
  }
  __syncthreads();

  // ---- Stage 2: row softmax (16 threads/row, width-16 shuffle reductions)
  {
    const int row = tid >> 4;
    const int sub = tid & 15;
    float* r = sc + row * 2048;
    float m = -3.0e38f;
    for (int c = sub; c < 2048; c += 16) m = fmaxf(m, r[c]);
#pragma unroll
    for (int o = 8; o >= 1; o >>= 1) m = fmaxf(m, __shfl_xor(m, o, 16));
    float s = 0.0f;
    for (int c = sub; c < 2048; c += 16) {
      const float e = __expf(r[c] - m);
      r[c] = e;
      s += e;
    }
#pragma unroll
    for (int o = 8; o >= 1; o >>= 1) s += __shfl_xor(s, o, 16);
    const float inv = 1.0f / s;
    for (int c = sub; c < 2048; c += 16) r[c] *= inv;
  }
  __syncthreads();

  // ---- Stage 2b: write attn stripe (16 contiguous rows -> flat 128KB, f32)
  {
    float4* dst = (float4*)(attn_out + (((size_t)b * 16 + h) * 2048 + q0) * 2048);
    for (int i = tid; i < 16 * 2048 / 4; i += 256)
      dst[i] = *(float4*)&sc[i * 4];
  }

  // ---- Stage 3: PV split-K: wave w covers k in [w*256, (w+1)*256)
  v8f pv[4] = {};
  for (int t = 0; t < 8; ++t) {
    const int kb = wave * 256 + t * 32;
    // A fragment: attn probs from LDS, f32 -> f16
    v16h afr;
    {
      const float* p = &sc[(lane & 15) * 2048 + kb + ((lane >> 4) << 3)];
#pragma unroll
      for (int j = 0; j < 8; ++j) afr[j] = (_Float16)p[j];
#pragma unroll
      for (int j = 0; j < 8; ++j) afr[8 + j] = (_Float16)p[16 + j];
    }
#pragma unroll
    for (int nt = 0; nt < 4; ++nt) {
      // B fragment: V[k, d] — lane's column d = nt*16 + cn, K strided by ldq.
      v16h bfr;
      const _Float16* vp = vbase + (size_t)kb * ldq + nt * 16 + cn;
#pragma unroll
      for (int j = 0; j < 8; ++j)
        bfr[j] = vp[(size_t)(rbase + j) * ldq];
#pragma unroll
      for (int j = 0; j < 8; ++j)
        bfr[8 + j] = vp[(size_t)(16 + rbase + j) * ldq];
      pv[nt] = WMMA_F16(afr, bfr, pv[nt]);
    }
  }

  // ---- Stage 4: reduce split-K partials across the 8 waves via LDS.
#pragma unroll
  for (int nt = 0; nt < 4; ++nt)
#pragma unroll
    for (int r = 0; r < 8; ++r)
      part[wave * 1024 + (rbase + r) * 64 + nt * 16 + cn] = pv[nt][r];
  __syncthreads();

  for (int i = tid; i < 1024; i += 256) {
    float s = 0.0f;
#pragma unroll
    for (int w = 0; w < 8; ++w) s += part[w * 1024 + i];
    const int r = i >> 6;
    const int d = i & 63;
    vals[((size_t)b * 2048 + q0 + r) * 1024 + h * 64 + d] = (_Float16)s;
  }
}

// ---------------------------------------------------------------------------
extern "C" void kernel_launch(void* const* d_in, const int* in_sizes, int n_in,
                              void* d_out, int out_size, void* d_ws, size_t ws_size,
                              hipStream_t stream) {
  (void)in_sizes; (void)n_in; (void)out_size; (void)ws_size;

  const float* x     = (const float*)d_in[0];  // [2,2048,1024]
  const float* w_qkv = (const float*)d_in[1];  // [3072,1024]
  const float* b_qkv = (const float*)d_in[2];  // [3072]
  const float* w_out = (const float*)d_in[3];  // [1024,1024]
  const float* b_out = (const float*)d_in[4];  // [1024]

  float* out  = (float*)d_out;                           // 4096*1024 f32
  float* attn = out + (size_t)2 * 2048 * 1024;           // 2*16*2048*2048 f32

  // Workspace layout (f16), total ~48MB:
  _Float16* xh    = (_Float16*)d_ws;                     // 4096*1024   (8MB)
  _Float16* wqkvh = xh + (size_t)4096 * 1024;            // 3072*1024   (6MB)
  _Float16* wouth = wqkvh + (size_t)3072 * 1024;         // 1024*1024   (2MB)
  _Float16* qkvh  = wouth + (size_t)1024 * 1024;         // 4096*3072  (24MB)
  _Float16* valsh = qkvh + (size_t)4096 * 3072;          // 4096*1024   (8MB)

  // 0) one-shot f32 -> f16 conversions of all GEMM operands
  {
    const int n4x = 4096 * 1024 / 4, n4q = 3072 * 1024 / 4, n4o = 1024 * 1024 / 4;
    cvt_f32_to_f16<<<(n4x + 255) / 256, 256, 0, stream>>>(x, xh, n4x);
    cvt_f32_to_f16<<<(n4q + 255) / 256, 256, 0, stream>>>(w_qkv, wqkvh, n4q);
    cvt_f32_to_f16<<<(n4o + 255) / 256, 256, 0, stream>>>(w_out, wouth, n4o);
  }

  // 1) qkv(f16) = x @ w_qkv^T + b_qkv   (M=4096, N=3072, K=1024)
  gemm_wmma<_Float16><<<dim3(4096 / 256, 3072 / 128), 256, 0, stream>>>(
      xh, wqkvh, b_qkv, qkvh, 4096, 3072, 1024);

  // 2) fused scores -> softmax -> attn write -> PV  (block per (b,h,qtile))
  const size_t smem_bytes = (16 * 2048 + 8 * 1024) * sizeof(float);  // 160KB
  attn_fused<<<dim3(2 * 16 * 128), 256, smem_bytes, stream>>>(qkvh, attn, valsh);

  // 3) out(f32) = vals(f16) @ w_out^T + b_out   (M=4096, N=1024, K=1024)
  gemm_wmma<float><<<dim3(4096 / 256, 1024 / 128), 256, 0, stream>>>(
      valsh, wouth, b_out, out, 4096, 1024, 1024);
}
